// MHA_62964220559535
// MI455X (gfx1250) — matrix operand order repaired and checked by reading, not compiled, not run on hardware
//
#include <hip/hip_runtime.h>
#include <hip/hip_bf16.h>
#include <math.h>

// ---------------- problem dims (compile-time) ----------------
constexpr int B_ = 2, S_ = 2048, D_ = 512, H_ = 8;

// ---------------- vector types ----------------
typedef __attribute__((ext_vector_type(16))) __bf16        v16bf;
typedef __attribute__((ext_vector_type(8)))  float         v8f;
typedef __attribute__((ext_vector_type(4)))  float         f32x4;
typedef __attribute__((ext_vector_type(4)))  unsigned int  u32x4;
typedef __attribute__((ext_vector_type(2)))  unsigned int  u32x2;
typedef __attribute__((ext_vector_type(8)))  int           i32x8;
typedef __attribute__((ext_vector_type(4)))  int           i32x4;

#define AS1 __attribute__((address_space(1)))
#define AS3 __attribute__((address_space(3)))

// ---- CDNA5 async-tensor feature detection (device pass only; safe fallback) ----
#if defined(__HIP_DEVICE_COMPILE__) && __has_builtin(__builtin_amdgcn_tensor_load_to_lds) && __has_builtin(__builtin_amdgcn_s_wait_tensorcnt)
#define HAVE_TDM 1
#else
#define HAVE_TDM 0
#endif

#if defined(__HIP_DEVICE_COMPILE__) && __has_builtin(__builtin_amdgcn_global_load_async_to_lds_b128)
#define HAVE_ASYNC 1
#else
#define HAVE_ASYNC 0
#endif

union FragU {
    u32x4 u[2];
    v16bf v;
    __bf16 h[16];
};

union PkU {
    u32x2 u;
    unsigned short s[4];
    __bf16 h[4];
};

// load a v16bf WMMA fragment from two 16-byte chunks
static __device__ __forceinline__ v16bf load_frag16(const __bf16* p0, const __bf16* p1) {
    FragU f;
    f.u[0] = *reinterpret_cast<const u32x4*>(p0);
    f.u[1] = *reinterpret_cast<const u32x4*>(p1);
    return f.v;
}

// ---------------- fp32 -> bf16 weight conversion ----------------
__global__ void conv_f32_bf16(const float* __restrict__ src, __bf16* __restrict__ dst, int n4) {
    int i = blockIdx.x * blockDim.x + threadIdx.x;
    if (i >= n4) return;
    f32x4 f = *reinterpret_cast<const f32x4*>(src + i * 4);
    __bf16* d = dst + i * 4;
    d[0] = (__bf16)f.x; d[1] = (__bf16)f.y; d[2] = (__bf16)f.z; d[3] = (__bf16)f.w;
}

// ---------------- GEMM tiling constants ----------------
constexpr int BM = 128, BN = 64, BK = 32;
constexpr int LDAS = 40;  // padded bf16 row stride (80B: 16B-aligned, bank-spread)
constexpr int LDBS = 40;

// ---------------- per-head projection GEMM ----------------
// Y[b,h,s,:] = X[b,s,:] @ W[h] + bias[h]   (X fp32, W bf16, Y bf16)
__global__ __launch_bounds__(256) void proj_gemm(
    const float* __restrict__ X,    // [B,S,D]
    const __bf16* __restrict__ W,   // [H,D,D] bf16
    const float* __restrict__ bias, // [H,D]
    __bf16* __restrict__ Y)         // [B,H,S,D] bf16
{
    __shared__ __bf16 sA[BM * LDAS];
    __shared__ __bf16 sB[BN * LDBS];   // stored transposed: sB[col][k]

    const int bh = blockIdx.z;
    const int b = bh / H_, h = bh % H_;
    const float*  A  = X    + (size_t)b * S_ * D_;
    const __bf16* Wp = W    + (size_t)h * D_ * D_;
    const float*  bp = bias + (size_t)h * D_;
    __bf16*       Yp = Y    + (size_t)bh * S_ * D_;

    const int m0 = blockIdx.x * BM, n0 = blockIdx.y * BN;
    const int tid = threadIdx.x, w = tid >> 5, lane = tid & 31;
    const int lrow = lane & 15, lhalf = lane >> 4;

    const int arow = tid >> 1, acb = (tid & 1) * 16;   // A: 128x32, 16/thread
    const int bkp = tid & 15, bcg = tid >> 4;          // B: k-pair 2*bkp, cols 4*bcg..4*bcg+3

    v8f acc[4] = {};

    for (int k0 = 0; k0 < D_; k0 += BK) {
        // ---- stage A tile (fp32 -> bf16) ----
        const float* ga = A + (size_t)(m0 + arow) * D_ + k0 + acb;
        __builtin_prefetch(ga + BK, 0, 1);   // global_prefetch next slab
        f32x4 f0 = *reinterpret_cast<const f32x4*>(ga + 0);
        f32x4 f1 = *reinterpret_cast<const f32x4*>(ga + 4);
        f32x4 f2 = *reinterpret_cast<const f32x4*>(ga + 8);
        f32x4 f3 = *reinterpret_cast<const f32x4*>(ga + 12);
        FragU fr;
        fr.h[0] = (__bf16)f0.x;  fr.h[1] = (__bf16)f0.y;  fr.h[2]  = (__bf16)f0.z;  fr.h[3]  = (__bf16)f0.w;
        fr.h[4] = (__bf16)f1.x;  fr.h[5] = (__bf16)f1.y;  fr.h[6]  = (__bf16)f1.z;  fr.h[7]  = (__bf16)f1.w;
        fr.h[8] = (__bf16)f2.x;  fr.h[9] = (__bf16)f2.y;  fr.h[10] = (__bf16)f2.z;  fr.h[11] = (__bf16)f2.w;
        fr.h[12] = (__bf16)f3.x; fr.h[13] = (__bf16)f3.y; fr.h[14] = (__bf16)f3.z;  fr.h[15] = (__bf16)f3.w;
        *reinterpret_cast<u32x4*>(&sA[arow * LDAS + acb])     = fr.u[0];
        *reinterpret_cast<u32x4*>(&sA[arow * LDAS + acb + 8]) = fr.u[1];

        // ---- stage B tile transposed, pack 2 K per dword -> ds_store_b32 ----
        const __bf16* gw0 = Wp + (size_t)(k0 + 2 * bkp) * D_ + n0 + 4 * bcg;
        __builtin_prefetch(gw0 + (size_t)BK * D_, 0, 1);
        PkU r0, r1;
        r0.u = *reinterpret_cast<const u32x2*>(gw0);
        r1.u = *reinterpret_cast<const u32x2*>(gw0 + D_);
        #pragma unroll
        for (int j = 0; j < 4; ++j) {
            unsigned pk = (unsigned)r0.s[j] | ((unsigned)r1.s[j] << 16);
            *reinterpret_cast<unsigned*>(&sB[(4 * bcg + j) * LDBS + 2 * bkp]) = pk;
        }

        __syncthreads();

        // ---- compute: one A-fragment, 4 WMMAs across N ----
        const __bf16* arp = &sA[(w * 16 + lrow) * LDAS + lhalf * 8];
        v16bf af = load_frag16(arp, arp + 16);
        #pragma unroll
        for (int n = 0; n < 4; ++n) {
            const __bf16* brp = &sB[(n * 16 + lrow) * LDBS + lhalf * 8];
            v16bf bf = load_frag16(brp, brp + 16);
            acc[n] = __builtin_amdgcn_wmma_f32_16x16x32_bf16(
                false, af, false, bf, (short)0, acc[n], false, false);
        }
        __syncthreads();
    }

    // ---- epilogue: bias + bf16 store ----
    #pragma unroll
    for (int n = 0; n < 4; ++n) {
        const int col = n0 + n * 16 + lrow;
        const float bv = bp[col];
        #pragma unroll
        for (int v = 0; v < 8; ++v) {
            const int row = m0 + w * 16 + lhalf * 8 + v;
            Yp[(size_t)row * D_ + col] = (__bf16)(acc[n][v] + bv);
        }
    }
}

// ---------------- attention row-sum kernel ----------------
// scores = q@k^T / sqrt(S); online softmax; rowscale = sum_t probs (fp ~1).
// K-tile staging: TDM tensor_load_to_lds (wave 0) -> async-to-LDS -> manual.
constexpr int KLDS = 528;   // 512 + 16 pad: 1056B rows (TDM pad: 8 dwords / 256 dwords)

__global__ __launch_bounds__(256) void attn_rowsum(
    const __bf16* __restrict__ qb,   // [B,H,S,D]
    const __bf16* __restrict__ kb,   // [B,H,S,D]
    float* __restrict__ scale)       // [B,H,S]
{
    __shared__ __bf16 sK[16 * KLDS];

    const int bh = blockIdx.y;
    const __bf16* qp = qb + (size_t)bh * S_ * D_;
    const __bf16* kp = kb + (size_t)bh * S_ * D_;

    const int tid = threadIdx.x, w = tid >> 5, lane = tid & 31;
    const int lrow = lane & 15, lhalf = lane >> 4;
    const int qr0 = blockIdx.x * 128 + w * 16;

    // preload this wave's 16x512 q block as 16 WMMA A-fragments
    v16bf qf[16];
    const __bf16* qrow = qp + (size_t)(qr0 + lrow) * D_;
    #pragma unroll
    for (int kk = 0; kk < 16; ++kk) {
        const __bf16* p0 = qrow + kk * 32 + lhalf * 8;
        qf[kk] = load_frag16(p0, p0 + 16);
    }

    float m[8], Z[8];
    #pragma unroll
    for (int v = 0; v < 8; ++v) { m[v] = -1e30f; Z[v] = 0.0f; }
    const float sc = 0.02209708691207961f;  // 1/sqrt(2048)

    for (int j = 0; j < S_ / 16; ++j) {
#if HAVE_TDM
        // ---- Tensor Data Mover: one DMA of the 16x512 bf16 tile, LDS-padded ----
        if (w == 0) {
            const unsigned lds_base = (unsigned)(size_t)(&sK[0]);
            const unsigned long long ga =
                (unsigned long long)(size_t)(kp + (size_t)(j * 16) * D_);
            u32x4 g0;
            g0.x = 1u;                                         // count=1, user mode
            g0.y = lds_base;                                   // lds_addr
            g0.z = (unsigned)ga;                               // global_addr[31:0]
            g0.w = (unsigned)((ga >> 32) & 0x1FFFFFFu) | (2u << 30);  // addr[56:32], type=2
            i32x8 g1 = {};
            // data_size=2B(code1), pad_enable, pad_interval=256dw(code7), pad_amount=8dw(code7)
            g1[0] = (int)((1u << 16) | (1u << 20) | (7u << 22) | (7u << 25));
            g1[1] = (int)((unsigned)D_ << 16);                 // tensor_dim0 lo16
            g1[2] = (int)(((unsigned)D_ >> 16) | ((unsigned)S_ << 16)); // dim0 hi | dim1 lo
            g1[3] = (int)(((unsigned)S_ >> 16) | ((unsigned)D_ << 16)); // dim1 hi | tile_dim0
            g1[4] = 16;                                        // tile_dim1=16, tile_dim2=0
            g1[5] = D_;                                        // tensor_dim0_stride lo32
            g1[6] = 0;
            g1[7] = 0;
            i32x4 gz = {};
#if __clang_major__ >= 23
            i32x8 gz8 = {};
            __builtin_amdgcn_tensor_load_to_lds(g0, g1, gz, gz, gz8, 0);
#else
            __builtin_amdgcn_tensor_load_to_lds(g0, g1, gz, gz, 0);
#endif
            __builtin_amdgcn_s_wait_tensorcnt(0);
        }
        __syncthreads();
#elif HAVE_ASYNC
        // ---- async copy global -> LDS (ASYNCcnt), 16B per lane per op ----
        #pragma unroll
        for (int c = 0; c < 4; ++c) {
            const int chunk = tid * 4 + c;
            const int kr = chunk >> 6, doff = (chunk & 63) * 8;
            __builtin_amdgcn_global_load_async_to_lds_b128(
                (AS1 i32x4*)(kp + (size_t)(j * 16 + kr) * D_ + doff),
                (AS3 i32x4*)(&sK[kr * KLDS + doff]), 0, 0);
        }
#if __has_builtin(__builtin_amdgcn_s_wait_asynccnt)
        __builtin_amdgcn_s_wait_asynccnt(0);
#else
        asm volatile("s_wait_asynccnt 0x0" ::: "memory");
#endif
        __syncthreads();
#else
        // ---- manual staging fallback ----
        #pragma unroll
        for (int c = 0; c < 4; ++c) {
            const int chunk = tid * 4 + c;
            const int kr = chunk >> 6, doff = (chunk & 63) * 8;
            *reinterpret_cast<u32x4*>(&sK[kr * KLDS + doff]) =
                *reinterpret_cast<const u32x4*>(kp + (size_t)(j * 16 + kr) * D_ + doff);
        }
        __syncthreads();
#endif

        // 16x16 score tile: 16 WMMAs over K=512
        v8f acc = {};
        #pragma unroll
        for (int kk = 0; kk < 16; ++kk) {
            const __bf16* bp = &sK[lrow * KLDS + kk * 32 + lhalf * 8];
            v16bf bf = load_frag16(bp, bp + 16);
            acc = __builtin_amdgcn_wmma_f32_16x16x32_bf16(
                false, qf[kk], false, bf, (short)0, acc, false, false);
        }

        // online softmax stats (reduce over 16 cols = 16 lanes per half-wave)
        #pragma unroll
        for (int v = 0; v < 8; ++v) {
            float s = acc[v] * sc;
            float tmax = s;
            #pragma unroll
            for (int mlt = 1; mlt < 16; mlt <<= 1)
                tmax = fmaxf(tmax, __shfl_xor(tmax, mlt, 16));
            const float mn = fmaxf(m[v], tmax);
            float p = expf(s - mn);
            #pragma unroll
            for (int mlt = 1; mlt < 16; mlt <<= 1)
                p += __shfl_xor(p, mlt, 16);
            Z[v] = Z[v] * expf(m[v] - mn) + p;
            m[v] = mn;
        }
        __syncthreads();
    }

    if (lrow == 0) {
        #pragma unroll
        for (int v = 0; v < 8; ++v) {
            const int row = qr0 + lhalf * 8 + v;
            scale[(size_t)bh * S_ + row] = Z[v] / Z[v];   // faithful softmax row-sum
        }
    }
}

// ---------------- scale v by rowsum and concat heads ----------------
__global__ void scale_concat(const __bf16* __restrict__ vb,
                             const float* __restrict__ scale,
                             __bf16* __restrict__ cat)
{
    const int c = blockIdx.x * blockDim.x + threadIdx.x;   // 8 bf16 per thread
    const int bh  = c / (S_ * (D_ / 8));
    const int rem = c - bh * (S_ * (D_ / 8));
    const int s   = rem >> 6;
    const int d0  = (rem & 63) * 8;
    const int b = bh / H_, h = bh % H_;

    const float sv = scale[(size_t)bh * S_ + s];
    FragU x;
    x.u[0] = *reinterpret_cast<const u32x4*>(vb + ((size_t)bh * S_ + s) * D_ + d0);
    #pragma unroll
    for (int i = 0; i < 8; ++i) x.h[i] = (__bf16)((float)x.h[i] * sv);
    *reinterpret_cast<u32x4*>(cat + (size_t)(b * S_ + s) * (H_ * D_) + h * D_ + d0) = x.u[0];
}

// ---------------- output projection GEMM ----------------
// Out[m,:] = cat[m,:] @ Wo + bo   (cat bf16 [4096,4096], Wo bf16 [4096,512], Out fp32)
__global__ __launch_bounds__(256) void out_gemm(
    const __bf16* __restrict__ Acat,
    const __bf16* __restrict__ W,
    const float* __restrict__ bias,
    float* __restrict__ Out)
{
    constexpr int KTOT = H_ * D_;     // 4096
    __shared__ __bf16 sA[BM * LDAS];
    __shared__ __bf16 sB[BN * LDBS];

    const int m0 = blockIdx.x * BM, n0 = blockIdx.y * BN;
    const int tid = threadIdx.x, w = tid >> 5, lane = tid & 31;
    const int lrow = lane & 15, lhalf = lane >> 4;

    const int arow = tid >> 1, acb = (tid & 1) * 16;
    const int bkp = tid & 15, bcg = tid >> 4;

    v8f acc[4] = {};

    for (int k0 = 0; k0 < KTOT; k0 += BK) {
        const __bf16* ga = Acat + (size_t)(m0 + arow) * KTOT + k0 + acb;
        __builtin_prefetch(ga + BK, 0, 1);
#if HAVE_ASYNC
        // A tile is already bf16: async copy straight into LDS (ASYNCcnt path)
        __builtin_amdgcn_global_load_async_to_lds_b128(
            (AS1 i32x4*)(ga), (AS3 i32x4*)(&sA[arow * LDAS + acb]), 0, 0);
        __builtin_amdgcn_global_load_async_to_lds_b128(
            (AS1 i32x4*)(ga + 8), (AS3 i32x4*)(&sA[arow * LDAS + acb + 8]), 0, 0);
#else
        *reinterpret_cast<u32x4*>(&sA[arow * LDAS + acb]) =
            *reinterpret_cast<const u32x4*>(ga);
        *reinterpret_cast<u32x4*>(&sA[arow * LDAS + acb + 8]) =
            *reinterpret_cast<const u32x4*>(ga + 8);
#endif

        const __bf16* gw0 = W + (size_t)(k0 + 2 * bkp) * D_ + n0 + 4 * bcg;
        __builtin_prefetch(gw0 + (size_t)BK * D_, 0, 1);
        PkU r0, r1;
        r0.u = *reinterpret_cast<const u32x2*>(gw0);
        r1.u = *reinterpret_cast<const u32x2*>(gw0 + D_);
        #pragma unroll
        for (int j = 0; j < 4; ++j) {
            unsigned pk = (unsigned)r0.s[j] | ((unsigned)r1.s[j] << 16);
            *reinterpret_cast<unsigned*>(&sB[(4 * bcg + j) * LDBS + 2 * bkp]) = pk;
        }

#if HAVE_ASYNC
#if __has_builtin(__builtin_amdgcn_s_wait_asynccnt)
        __builtin_amdgcn_s_wait_asynccnt(0);
#else
        asm volatile("s_wait_asynccnt 0x0" ::: "memory");
#endif
#endif
        __syncthreads();

        const __bf16* arp = &sA[(w * 16 + lrow) * LDAS + lhalf * 8];
        v16bf af = load_frag16(arp, arp + 16);
        #pragma unroll
        for (int n = 0; n < 4; ++n) {
            const __bf16* brp = &sB[(n * 16 + lrow) * LDBS + lhalf * 8];
            v16bf bf = load_frag16(brp, brp + 16);
            acc[n] = __builtin_amdgcn_wmma_f32_16x16x32_bf16(
                false, af, false, bf, (short)0, acc[n], false, false);
        }
        __syncthreads();
    }

    #pragma unroll
    for (int n = 0; n < 4; ++n) {
        const int col = n0 + n * 16 + lrow;
        const float bv = bias[col];
        #pragma unroll
        for (int v = 0; v < 8; ++v) {
            const int row = m0 + w * 16 + lhalf * 8 + v;
            Out[(size_t)row * D_ + col] = acc[n][v] + bv;
        }
    }
}

// ---------------- host-side orchestration ----------------
extern "C" void kernel_launch(void* const* d_in, const int* in_sizes, int n_in,
                              void* d_out, int out_size, void* d_ws, size_t ws_size,
                              hipStream_t stream) {
    (void)in_sizes; (void)n_in; (void)out_size; (void)ws_size;
    // input order: Q K V Wq bq Wk bk Wv bv Wo bo
    const float* Q  = (const float*)d_in[0];
    const float* K  = (const float*)d_in[1];
    const float* V  = (const float*)d_in[2];
    const float* Wq = (const float*)d_in[3];
    const float* bq = (const float*)d_in[4];
    const float* Wk = (const float*)d_in[5];
    const float* bk = (const float*)d_in[6];
    const float* Wv = (const float*)d_in[7];
    const float* bv = (const float*)d_in[8];
    const float* Wo = (const float*)d_in[9];
    const float* bo = (const float*)d_in[10];
    float* Out = (float*)d_out;

    // workspace layout (bytes)
    char* ws = (char*)d_ws;
    const size_t szWh  = (size_t)H_ * D_ * D_ * sizeof(__bf16);      // 4 MB
    const size_t szWo  = (size_t)(H_ * D_) * D_ * sizeof(__bf16);    // 4 MB
    const size_t szQKV = (size_t)B_ * H_ * S_ * D_ * sizeof(__bf16); // 32 MB
    const size_t szScl = (size_t)B_ * H_ * S_ * sizeof(float);
    __bf16* Wq_b = (__bf16*)(ws);
    __bf16* Wk_b = (__bf16*)(ws + szWh);
    __bf16* Wv_b = (__bf16*)(ws + 2 * szWh);
    __bf16* Wo_b = (__bf16*)(ws + 3 * szWh);
    __bf16* q_b  = (__bf16*)(ws + 3 * szWh + szWo);
    __bf16* k_b  = (__bf16*)(ws + 3 * szWh + szWo + szQKV);
    __bf16* v_b  = (__bf16*)(ws + 3 * szWh + szWo + 2 * szQKV);
    float*  scl  = (float*) (ws + 3 * szWh + szWo + 3 * szQKV);
    __bf16* cat  = (__bf16*)(ws + 3 * szWh + szWo + 3 * szQKV + szScl);

    // 1) weights fp32 -> bf16
    const int nW4 = H_ * D_ * D_ / 4;       // 524288
    conv_f32_bf16<<<nW4 / 256, 256, 0, stream>>>(Wq, Wq_b, nW4);
    conv_f32_bf16<<<nW4 / 256, 256, 0, stream>>>(Wk, Wk_b, nW4);
    conv_f32_bf16<<<nW4 / 256, 256, 0, stream>>>(Wv, Wv_b, nW4);
    conv_f32_bf16<<<nW4 / 256, 256, 0, stream>>>(Wo, Wo_b, nW4);

    // 2) per-head projections (WMMA GEMMs)
    dim3 gProj(S_ / BM, D_ / BN, B_ * H_);   // (16,8,16)
    proj_gemm<<<gProj, 256, 0, stream>>>(Q, Wq_b, bq, q_b);
    proj_gemm<<<gProj, 256, 0, stream>>>(K, Wk_b, bk, k_b);
    proj_gemm<<<gProj, 256, 0, stream>>>(V, Wv_b, bv, v_b);

    // 3) attention scores + softmax row-sums (WMMA q@k^T, TDM/async staging)
    dim3 gAttn(S_ / 128, B_ * H_);           // (16,16)
    attn_rowsum<<<gAttn, 256, 0, stream>>>(q_b, k_b, scl);

    // 4) scale v and concat heads
    const int nChunks = B_ * H_ * S_ * (D_ / 8);   // 2097152
    scale_concat<<<nChunks / 256, 256, 0, stream>>>(v_b, scl, cat);

    // 5) output projection (WMMA GEMM, K=4096) -> fp32
    dim3 gOut((B_ * S_) / BM, D_ / BN);      // (32,8)
    out_gemm<<<gOut, 256, 0, stream>>>(cat, Wo_b, bo, Out);
}
